// AttentionBlock_11725260718152
// MI455X (gfx1250) — matrix operand order repaired and checked by reading, not compiled, not run on hardware
//
#include <hip/hip_runtime.h>

// ---------------------------------------------------------------------------
// Problem constants (match reference)
// ---------------------------------------------------------------------------
#define Bn        8
#define Cn        512
#define Hn        32
#define Wn        32
#define Nn        (Hn * Wn)      // 1024
#define NHEADS    8
#define HEADDIM   64             // C / NHEADS
#define GROUPS    32
#define CPG       (Cn / GROUPS)  // 16 channels per group
#define EPSG      1e-5f
// scale = 1/sqrt(sqrt(head_dim)) = 64^-0.25
#define QK_SCALE  0.35355339059327373f

typedef __attribute__((ext_vector_type(16))) _Float16 v16h;
typedef __attribute__((ext_vector_type(8)))  _Float16 v8h;
typedef __attribute__((ext_vector_type(8)))  float    v8f;

// ---------------------------------------------------------------------------
// WMMA helpers (wave32, V_WMMA_F32_16X16X32_F16)
//
// A fragment: 16x32 (MxK), source row-major MxK (contiguous K).
//   lane l: m = l&15 ; k = (l>>4)*16 + j   (two contiguous 8-half = 16B loads)
// B fragment: 32x16 (KxN), source stored TRANSPOSED, i.e. row-major NxK.
//   lane l: n = l&15 ; k = (l>>4)*16 + j  (one contiguous 16-half = 32B load)
// C/D: 8 VGPRs, lane l: n = l&15 ; m = r + (l>>4)*8
// ---------------------------------------------------------------------------
__device__ __forceinline__ v16h frag_a(const _Float16* p, int ld, int m0, int k0) {
  const int lane = threadIdx.x & 31;
  const _Float16* base = p + (size_t)(m0 + (lane & 15)) * ld + k0 + ((lane >> 4) << 4);
  v8h lo = *(const v8h*)(base);
  v8h hi = *(const v8h*)(base + 8);
  return __builtin_shufflevector(lo, hi, 0, 1, 2, 3, 4, 5, 6, 7,
                                 8, 9, 10, 11, 12, 13, 14, 15);
}

__device__ __forceinline__ v16h frag_bT(const _Float16* p, int ld, int k0, int n0) {
  const int lane = threadIdx.x & 31;
  return *(const v16h*)(p + (size_t)(n0 + (lane & 15)) * ld + k0 + ((lane >> 4) << 4));
}

__device__ __forceinline__ v8f wmma_f16(v16h a, v16h b, v8f c) {
  return __builtin_amdgcn_wmma_f32_16x16x32_f16(
      /*neg_a=*/false, a, /*neg_b=*/false, b,
      /*c_mod=*/(short)0, c, /*reuse_a=*/false, /*reuse_b=*/false);
}

__device__ __forceinline__ void store_tile_f32(float* dst, int ld, int m0, int n0, v8f c) {
  const int lane = threadIdx.x & 31;
  const int n = n0 + (lane & 15);
  const int mb = m0 + ((lane >> 4) << 3);
#pragma unroll
  for (int r = 0; r < 8; ++r) dst[(size_t)(mb + r) * ld + n] = c[r];
}

// ---------------------------------------------------------------------------
// Kernel 0: f32 -> f16 conversion (weights)
// ---------------------------------------------------------------------------
__global__ __launch_bounds__(256) void k_cvt_f16(const float* __restrict__ src,
                                                 _Float16* __restrict__ dst, int n) {
  int i = blockIdx.x * 256 + threadIdx.x;
  if (i < n) dst[i] = (_Float16)src[i];
}

// ---------------------------------------------------------------------------
// Kernel 1: GroupNorm. One block per (batch, group) = 256 blocks.
// Writes xnT as (b, n, c) f16 so QKV B-fragments are contiguous.
// ---------------------------------------------------------------------------
__global__ __launch_bounds__(256) void k_groupnorm(const float* __restrict__ x,
                                                   const float* __restrict__ gw,
                                                   const float* __restrict__ gb,
                                                   _Float16* __restrict__ xnT) {
  const int bg = blockIdx.x;
  const int b = bg / GROUPS;
  const int g = bg % GROUPS;
  const float* xp = x + ((size_t)b * Cn + g * CPG) * Nn;  // CPG x Nn block
  const int tid = threadIdx.x;

  float s = 0.f, s2 = 0.f;
  for (int i = tid; i < CPG * Nn; i += 256) {
    float v = xp[i];
    s += v; s2 += v * v;
  }
  __shared__ float rs[256], rs2[256];
  rs[tid] = s; rs2[tid] = s2;
  __syncthreads();
  for (int off = 128; off > 0; off >>= 1) {
    if (tid < off) { rs[tid] += rs[tid + off]; rs2[tid] += rs2[tid + off]; }
    __syncthreads();
  }
  const float inv_n = 1.0f / (float)(CPG * Nn);
  const float mean = rs[0] * inv_n;
  const float var = rs2[0] * inv_n - mean * mean;
  const float rstd = rsqrtf(var + EPSG);

  for (int i = tid; i < CPG * Nn; i += 256) {
    int ci = i >> 10;            // local channel
    int n  = i & (Nn - 1);
    int c  = g * CPG + ci;
    float y = (xp[i] - mean) * rstd * gw[c] + gb[c];
    xnT[((size_t)b * Nn + n) * Cn + c] = (_Float16)y;
  }
}

// ---------------------------------------------------------------------------
// Kernel 2: QKV GEMM.  Out(1536 x 1024) = W(1536x512) @ XnT^T per batch.
// grid (24, 8, B): block tile 64(M) x 128(N), 8 waves.
// Wave w: m-tile = w>>1, n-tiles = (w&1)*4 .. +3 (4 accumulators).
// Epilogue scatters into qT/kT (bh, n, c) [scaled] and V (bh, c, s).
// ---------------------------------------------------------------------------
__global__ __launch_bounds__(256) void k_qkv_gemm(const _Float16* __restrict__ w16,
                                                  const _Float16* __restrict__ xnT,
                                                  _Float16* __restrict__ qT,
                                                  _Float16* __restrict__ kT,
                                                  _Float16* __restrict__ vbuf) {
  const int b = blockIdx.z;
  const int wave = threadIdx.x >> 5;
  const int lane = threadIdx.x & 31;
  const int m0 = blockIdx.x * 64 + (wave >> 1) * 16;   // output channel tile
  const int n0b = blockIdx.y * 128 + (wave & 1) * 64;  // position tile base
  const _Float16* Bp = xnT + (size_t)b * Nn * Cn;      // NxK row-major, ld=Cn

  v8f acc[4] = {v8f{}, v8f{}, v8f{}, v8f{}};
  for (int k0 = 0; k0 < Cn; k0 += 32) {
    v16h a = frag_a(w16, Cn, m0, k0);
#pragma unroll
    for (int t = 0; t < 4; ++t) {
      v16h bf = frag_bT(Bp, Cn, k0, n0b + t * 16);
      acc[t] = wmma_f16(a, bf, acc[t]);
    }
  }

  const int mb = m0 + ((lane >> 4) << 3);  // 8 consecutive output channels
  const int h = mb / (3 * HEADDIM);
  const int rr = mb % (3 * HEADDIM);
  const int bh = b * NHEADS + h;
#pragma unroll
  for (int t = 0; t < 4; ++t) {
    const int n = n0b + t * 16 + (lane & 15);
    if (rr < HEADDIM) {                 // Q -> (bh, n, c), scaled
      v8h hv;
#pragma unroll
      for (int r = 0; r < 8; ++r) hv[r] = (_Float16)(acc[t][r] * QK_SCALE);
      *(v8h*)(qT + ((size_t)bh * Nn + n) * HEADDIM + rr) = hv;
    } else if (rr < 2 * HEADDIM) {      // K -> (bh, n, c), scaled
      v8h hv;
#pragma unroll
      for (int r = 0; r < 8; ++r) hv[r] = (_Float16)(acc[t][r] * QK_SCALE);
      *(v8h*)(kT + ((size_t)bh * Nn + n) * HEADDIM + (rr - HEADDIM)) = hv;
    } else {                            // V -> (bh, c, s)
#pragma unroll
      for (int r = 0; r < 8; ++r)
        vbuf[((size_t)bh * HEADDIM + (rr - 2 * HEADDIM) + r) * Nn + n] = (_Float16)acc[t][r];
    }
  }
}

// ---------------------------------------------------------------------------
// Kernel 3: fused attention.  grid (Nn/32, B*NHEADS), 256 threads (8 waves).
// Per block: one (bh) and a 32-row query tile.
//   Phase 1: S[32][1024] = (Q^T K) via WMMA -> LDS f32 (128 KB)
//   Phase 2: row softmax (8 threads per row), P -> LDS f16 (64 KB)
//   Phase 3: H(64c x 32t) = V @ P^T via WMMA, B-frags from LDS
// Dynamic LDS: 32*1024*4 + 32*1024*2 = 192 KB (of 320 KB per WGP).
// ---------------------------------------------------------------------------
#define ATTN_LDS_BYTES (32 * Nn * 4 + 32 * Nn * 2)

__global__ __launch_bounds__(256) void k_attn(const _Float16* __restrict__ qT,
                                              const _Float16* __restrict__ kT,
                                              const _Float16* __restrict__ vbuf,
                                              _Float16* __restrict__ hT) {
  extern __shared__ char smem[];
  float*    sS = (float*)smem;                       // [32][1024] scores
  _Float16* sP = (_Float16*)(smem + 32 * Nn * 4);    // [32][1024] probs f16
  __shared__ float red[32][8];

  const int bh = blockIdx.y;
  const int t0 = blockIdx.x * 32;
  const int wave = threadIdx.x >> 5;

  const _Float16* Qp = qT + (size_t)bh * Nn * HEADDIM;   // (t, c) ld=64
  const _Float16* Kp = kT + (size_t)bh * Nn * HEADDIM;   // (s, c) ld=64  == B^T
  const _Float16* Vp = vbuf + (size_t)bh * HEADDIM * Nn; // (c, s) ld=1024

  // ---- Phase 1: scores.  Wave w covers s in [w*128, w*128+128).
  {
    v16h a00 = frag_a(Qp, HEADDIM, t0 + 0, 0);
    v16h a01 = frag_a(Qp, HEADDIM, t0 + 0, 32);
    v16h a10 = frag_a(Qp, HEADDIM, t0 + 16, 0);
    v16h a11 = frag_a(Qp, HEADDIM, t0 + 16, 32);
#pragma unroll
    for (int si = 0; si < 8; ++si) {
      const int s0 = wave * 128 + si * 16;
      v16h b0 = frag_bT(Kp, HEADDIM, 0, s0);
      v16h b1 = frag_bT(Kp, HEADDIM, 32, s0);
      v8f c0 = {}; c0 = wmma_f16(a00, b0, c0); c0 = wmma_f16(a01, b1, c0);
      v8f c1 = {}; c1 = wmma_f16(a10, b0, c1); c1 = wmma_f16(a11, b1, c1);
      store_tile_f32(sS, Nn, 0, s0, c0);
      store_tile_f32(sS, Nn, 16, s0, c1);
    }
  }
  __syncthreads();

  // ---- Phase 2: softmax over s (1024) for each of 32 rows; 8 thr/row.
  {
    const int row = threadIdx.x >> 3;
    const int sub = threadIdx.x & 7;
    float* Sr = sS + row * Nn;
    float mx = -3.0e38f;
    for (int i = sub * 128; i < sub * 128 + 128; ++i) mx = fmaxf(mx, Sr[i]);
    red[row][sub] = mx;
    __syncthreads();
    mx = red[row][0];
#pragma unroll
    for (int j = 1; j < 8; ++j) mx = fmaxf(mx, red[row][j]);
    __syncthreads();
    float sum = 0.f;
    for (int i = sub * 128; i < sub * 128 + 128; ++i) {
      float e = __expf(Sr[i] - mx);
      Sr[i] = e;
      sum += e;
    }
    red[row][sub] = sum;
    __syncthreads();
    float tot = 0.f;
#pragma unroll
    for (int j = 0; j < 8; ++j) tot += red[row][j];
    const float inv = 1.0f / tot;
    for (int i = sub * 128; i < sub * 128 + 128; ++i)
      sP[row * Nn + i] = (_Float16)(Sr[i] * inv);
  }
  __syncthreads();

  // ---- Phase 3: H(64 x 32) = V(64 x 1024) @ P^T(1024 x 32).
  // 8 output tiles (4 c-tiles x 2 t-tiles) -> one per wave.
  {
    const int c0 = (wave >> 1) * 16;
    const int tt0 = (wave & 1) * 16;
    v8f acc = {};
    for (int k0 = 0; k0 < Nn; k0 += 32) {
      v16h a = frag_a(Vp, Nn, c0, k0);
      v16h bf = frag_bT(sP, Nn, k0, tt0);  // sP is (t, s) row-major == (P^T)^T
      acc = wmma_f16(a, bf, acc);
    }
    const int lane = threadIdx.x & 31;
    const int b = bh >> 3, hh = bh & 7;
    const int gn = t0 + tt0 + (lane & 15);
    const int cb = hh * HEADDIM + c0 + ((lane >> 4) << 3);
    v8h hv;
#pragma unroll
    for (int r = 0; r < 8; ++r) hv[r] = (_Float16)acc[r];
    *(v8h*)(hT + ((size_t)b * Nn + gn) * Cn + cb) = hv;  // (b, n, c)
  }
}

// ---------------------------------------------------------------------------
// Kernel 4: proj GEMM + bias + residual.
// Out(512 x 1024) = Pw(512x512) @ H + bias + x.  grid (8, 8, B).
// ---------------------------------------------------------------------------
__global__ __launch_bounds__(256) void k_proj_gemm(const _Float16* __restrict__ pw16,
                                                   const _Float16* __restrict__ hT,
                                                   const float* __restrict__ pb,
                                                   const float* __restrict__ x,
                                                   float* __restrict__ out) {
  const int b = blockIdx.z;
  const int wave = threadIdx.x >> 5;
  const int lane = threadIdx.x & 31;
  const int m0 = blockIdx.x * 64 + (wave >> 1) * 16;
  const int n0b = blockIdx.y * 128 + (wave & 1) * 64;
  const _Float16* Bp = hT + (size_t)b * Nn * Cn;  // NxK row-major, ld=Cn

  v8f acc[4] = {v8f{}, v8f{}, v8f{}, v8f{}};
  for (int k0 = 0; k0 < Cn; k0 += 32) {
    v16h a = frag_a(pw16, Cn, m0, k0);
#pragma unroll
    for (int t = 0; t < 4; ++t) {
      v16h bf = frag_bT(Bp, Cn, k0, n0b + t * 16);
      acc[t] = wmma_f16(a, bf, acc[t]);
    }
  }

  const int mb = m0 + ((lane >> 4) << 3);
#pragma unroll
  for (int t = 0; t < 4; ++t) {
    const int n = n0b + t * 16 + (lane & 15);
#pragma unroll
    for (int r = 0; r < 8; ++r) {
      const size_t idx = ((size_t)b * Cn + mb + r) * Nn + n;
      out[idx] = acc[t][r] + pb[mb + r] + x[idx];
    }
  }
}

// ---------------------------------------------------------------------------
// Host launcher
// ---------------------------------------------------------------------------
extern "C" void kernel_launch(void* const* d_in, const int* in_sizes, int n_in,
                              void* d_out, int out_size, void* d_ws, size_t ws_size,
                              hipStream_t stream) {
  (void)in_sizes; (void)n_in; (void)out_size; (void)ws_size;
  const float* x      = (const float*)d_in[0];
  const float* norm_w = (const float*)d_in[1];
  const float* norm_b = (const float*)d_in[2];
  const float* qkv_w  = (const float*)d_in[3];
  const float* proj_w = (const float*)d_in[4];
  const float* proj_b = (const float*)d_in[5];
  float* out = (float*)d_out;

  // Workspace carve-up (all f16 buffers; ~44 MB total)
  char* ws = (char*)d_ws;
  _Float16* w16q = (_Float16*)ws;                 ws += (size_t)3 * Cn * Cn * 2;   // 1536x512
  _Float16* w16p = (_Float16*)ws;                 ws += (size_t)Cn * Cn * 2;       // 512x512
  _Float16* xnT  = (_Float16*)ws;                 ws += (size_t)Bn * Nn * Cn * 2;  // (b,n,c)
  _Float16* qT   = (_Float16*)ws;                 ws += (size_t)Bn * NHEADS * Nn * HEADDIM * 2;
  _Float16* kT   = (_Float16*)ws;                 ws += (size_t)Bn * NHEADS * Nn * HEADDIM * 2;
  _Float16* vb   = (_Float16*)ws;                 ws += (size_t)Bn * NHEADS * HEADDIM * Nn * 2;
  _Float16* hT   = (_Float16*)ws;                 ws += (size_t)Bn * Nn * Cn * 2;

  // 0) weight conversion
  k_cvt_f16<<<(3 * Cn * Cn + 255) / 256, 256, 0, stream>>>(qkv_w, w16q, 3 * Cn * Cn);
  k_cvt_f16<<<(Cn * Cn + 255) / 256, 256, 0, stream>>>(proj_w, w16p, Cn * Cn);

  // 1) GroupNorm -> xnT (b, n, c) f16
  k_groupnorm<<<Bn * GROUPS, 256, 0, stream>>>(x, norm_w, norm_b, xnT);

  // 2) QKV GEMM -> qT/kT (scaled) + V
  k_qkv_gemm<<<dim3(3 * Cn / 64, Nn / 128, Bn), 256, 0, stream>>>(w16q, xnT, qT, kT, vb);

  // 3) fused attention (192 KB dynamic LDS per workgroup)
  k_attn<<<dim3(Nn / 32, Bn * NHEADS), 256, ATTN_LDS_BYTES, stream>>>(qT, kT, vb, hT);

  // 4) proj GEMM + bias + residual
  k_proj_gemm<<<dim3(Cn / 64, Nn / 128, Bn), 256, 0, stream>>>(w16p, hT, proj_b, x, out);
}
